// MICA_Net_1400159339183
// MI455X (gfx1250) — matrix operand order repaired and checked by reading, not compiled, not run on hardware
//
#include <hip/hip_runtime.h>
#include <hip/hip_bf16.h>

// ---------------------------------------------------------------------------
// MICA-Net fused inference for MI455X (gfx1250), bf16 WMMA path.
//
// Math folding (exact):
//   imu_att[j] = tanh((Waa[j,0]*<imu,imu> + Waa[j,1]*<imu,vis>)/8)
//   vis_att[j] = tanh((Wav[j,0]*<imu,vis> + Wav[j,1]*<vis,vis>)/8)
//   H_a = relu(ALPHA   * [W_a | W_ca] @ [imu, imu_att])
//   H_v = relu((1-A)   * [W_v | W_cv] @ [vis, vis_att])
//   ha  = W_ha @ H_a ; hv = W_hv @ H_v
//   out = W_combo @ [imu(64), vis(64), ha(8), hv(8)] + b_combo
// where W_combo = W_r2 @ fold(W_r1) absorbs the broadcast-concat + both
// linear head layers (K reduced from 1024 to 144, padded to 160).
// ---------------------------------------------------------------------------

typedef __attribute__((ext_vector_type(16))) __bf16 v16bf;
typedef __attribute__((ext_vector_type(8)))  float  v8f;

#define ALPHA_F  0.8f
#define OMA_F    0.2f

// ---- ws layout (bytes) ----
#define WS_WFOLD   0          // float[128*160]
#define WS_W1B     81920      // bf16 [64*768]
#define WS_W2B     180224     // bf16 [64*608]  (cols 600..607 zero)
#define WS_WCATA   258048     // bf16 [32*96]   ([W_a | W_ca | 0])
#define WS_WCATV   264192     // bf16 [32*96]
#define WS_WHA     270336     // bf16 [16*32]   (rows 8..15 zero)
#define WS_WHV     271360     // bf16 [16*32]
#define WS_WCMB    272384     // bf16 [32*160]
#define WS_BCMB    282624     // float[32]

__device__ __forceinline__ v8f v8f_splat(float x) {
  v8f r;
#pragma unroll
  for (int i = 0; i < 8; ++i) r[i] = x;
  return r;
}

__device__ __forceinline__ v8f wmma_bf16(v16bf a, v16bf b, v8f c) {
  return __builtin_amdgcn_wmma_f32_16x16x32_bf16(false, a, false, b,
                                                 (short)0, c, false, false);
}

// B fragment: W row-major [N][ldk] bf16 -> tile of W^T at (n0,k0).
// ISA bf16 B layout: lane (0..15) = column n, K {k0..k0+7, k0+16..k0+23}; lanes 16..31 K {k0+8..k0+15, k0+24..k0+31}.
__device__ __forceinline__ v16bf loadB(const __bf16* W, int ldk, int n0, int k0) {
  typedef __attribute__((ext_vector_type(8))) __bf16 v8bf_t;
  const int lane = threadIdx.x & 31;
  const int n  = n0 + (lane & 15);
  const int kk = k0 + ((lane >> 4) & 1) * 8;
  const __bf16* p = W + (size_t)n * ldk + kk;
  v8bf_t lo = *(const v8bf_t*)(p);        // 16B, 16B-aligned
  v8bf_t hi = *(const v8bf_t*)(p + 16);
  return __builtin_shufflevector(lo, hi, 0, 1, 2, 3, 4, 5, 6, 7, 8, 9, 10, 11, 12, 13, 14, 15);
}

// A fragment from global f32 row (convert to bf16).
// ISA bf16 A layout: lanes 0-15 hold K {k0..k0+7, k0+16..k0+23},
// lanes 16-31 hold K {k0+8..k0+15, k0+24..k0+31}. kmax is a multiple of 8.
__device__ __forceinline__ v16bf loadA_gl(const float* rowbase, int k0, int kmax) {
  const int lh = (threadIdx.x >> 4) & 1;
  const int c1 = k0 + lh * 8;
  const int c2 = k0 + 16 + lh * 8;
  float f[16];
  if (c1 < kmax) {
    float4 x = *(const float4*)(rowbase + c1);
    float4 y = *(const float4*)(rowbase + c1 + 4);
    f[0]=x.x; f[1]=x.y; f[2]=x.z; f[3]=x.w; f[4]=y.x; f[5]=y.y; f[6]=y.z; f[7]=y.w;
  } else {
#pragma unroll
    for (int i = 0; i < 8; ++i) f[i] = 0.f;
  }
  if (c2 < kmax) {
    float4 x = *(const float4*)(rowbase + c2);
    float4 y = *(const float4*)(rowbase + c2 + 4);
    f[8]=x.x; f[9]=x.y; f[10]=x.z; f[11]=x.w; f[12]=y.x; f[13]=y.y; f[14]=y.z; f[15]=y.w;
  } else {
#pragma unroll
    for (int i = 8; i < 16; ++i) f[i] = 0.f;
  }
  v16bf r;
#pragma unroll
  for (int i = 0; i < 16; ++i) r[i] = (__bf16)f[i];
  return r;
}

// A fragment from per-wave LDS f32 tile (row-major, given dword stride).
__device__ __forceinline__ v16bf loadA_lds(const float* base, int stride, int k0) {
  const int lane = threadIdx.x & 31;
  const float* p = base + (lane & 15) * stride + k0 + ((lane >> 4) & 1) * 8;
  v16bf r;
#pragma unroll
  for (int i = 0; i < 8; ++i) r[i] = (__bf16)p[i];
#pragma unroll
  for (int i = 0; i < 8; ++i) r[8 + i] = (__bf16)p[16 + i];
  return r;
}

// ---------------------------------------------------------------------------
// prep1: Wfold[n][col] (f32, [128][160]) from W_r1 [128][1024]
// col  0.. 63 : sum_r W_r1[n][r*128+col]                (imu broadcast terms)
// col 64..127 : sum_r W_r1[n][r*128+64+(col-64)]        (vis broadcast terms)
// col128..135 : ALPHA * sum_{c<64} W_r1[n][r*128+c]     (ha terms, r=col-128)
// col136..143 : (1-A) * sum_{c<64} W_r1[n][r*128+64+c]  (hv terms, r=col-136)
// col144..159 : 0   (K padding)
// ---------------------------------------------------------------------------
__global__ void prep_fold(const float* __restrict__ Wr1, float* __restrict__ Wfold) {
  int idx = blockIdx.x * blockDim.x + threadIdx.x;
  if (idx >= 128 * 160) return;
  int n = idx / 160, col = idx % 160;
  const float* w = Wr1 + (size_t)n * 1024;
  float s = 0.f;
  if (col < 64) {
#pragma unroll
    for (int r = 0; r < 8; ++r) s += w[r * 128 + col];
  } else if (col < 128) {
    int c = col - 64;
#pragma unroll
    for (int r = 0; r < 8; ++r) s += w[r * 128 + 64 + c];
  } else if (col < 136) {
    int r = col - 128;
    for (int c = 0; c < 64; ++c) s += w[r * 128 + c];
    s *= ALPHA_F;
  } else if (col < 144) {
    int r = col - 136;
    for (int c = 0; c < 64; ++c) s += w[r * 128 + 64 + c];
    s *= OMA_F;
  }
  Wfold[idx] = s;
}

// ---------------------------------------------------------------------------
// prep2: pack all bf16 weight operands + W_combo + b_combo into ws.
// ---------------------------------------------------------------------------
__global__ void prep_pack(const float* __restrict__ Wenc1, const float* __restrict__ Wenc2,
                          const float* __restrict__ Wa,    const float* __restrict__ Wv,
                          const float* __restrict__ Wca,   const float* __restrict__ Wcv,
                          const float* __restrict__ Wha,   const float* __restrict__ Whv,
                          const float* __restrict__ Wr2,   const float* __restrict__ br1,
                          const float* __restrict__ br2,   const float* __restrict__ Wfold,
                          __bf16* W1b, __bf16* W2b, __bf16* WcatA, __bf16* WcatV,
                          __bf16* Whab, __bf16* Whvb, __bf16* Wcmb, float* bcmb) {
  int i = blockIdx.x * blockDim.x + threadIdx.x;
  if (i < 64 * 768) { W1b[i] = (__bf16)Wenc1[i]; return; }
  i -= 64 * 768;
  if (i < 64 * 608) {
    int n = i / 608, c = i % 608;
    W2b[i] = (__bf16)((c < 600) ? Wenc2[n * 600 + c] : 0.f);
    return;
  }
  i -= 64 * 608;
  if (i < 32 * 96) {
    int n = i / 96, c = i % 96;
    float v = (c < 64) ? Wa[n * 64 + c] : ((c < 72) ? Wca[n * 8 + (c - 64)] : 0.f);
    WcatA[i] = (__bf16)v;
    return;
  }
  i -= 32 * 96;
  if (i < 32 * 96) {
    int n = i / 96, c = i % 96;
    float v = (c < 64) ? Wv[n * 64 + c] : ((c < 72) ? Wcv[n * 8 + (c - 64)] : 0.f);
    WcatV[i] = (__bf16)v;
    return;
  }
  i -= 32 * 96;
  if (i < 16 * 32) {
    int r = i / 32, c = i % 32;
    Whab[i] = (__bf16)((r < 8) ? Wha[r * 32 + c] : 0.f);
    return;
  }
  i -= 16 * 32;
  if (i < 16 * 32) {
    int r = i / 32, c = i % 32;
    Whvb[i] = (__bf16)((r < 8) ? Whv[r * 32 + c] : 0.f);
    return;
  }
  i -= 16 * 32;
  if (i < 32 * 160) {                 // W_combo = W_r2 @ Wfold
    int o = i / 160, c = i % 160;
    float s = 0.f;
    for (int n = 0; n < 128; ++n) s += Wr2[o * 128 + n] * Wfold[n * 160 + c];
    Wcmb[i] = (__bf16)s;
    return;
  }
  i -= 32 * 160;
  if (i < 32) {                       // b_combo = b_r2 + W_r2 @ b_r1
    float s = br2[i];
    for (int n = 0; n < 128; ++n) s += Wr2[i * 128 + n] * br1[n];
    bcmb[i] = s;
  }
}

// ---------------------------------------------------------------------------
// Main fused kernel: 2 waves / block, each wave owns 16 samples.
// ---------------------------------------------------------------------------
#define ST_IV 68   // 16B-aligned rows, bank stride 4
#define ST_H  36

__global__ __launch_bounds__(64) void mica_main(
    const float* __restrict__ f1, const float* __restrict__ f2,
    const float* __restrict__ benc1, const float* __restrict__ benc2,
    const float* __restrict__ Waa, const float* __restrict__ Wav,
    const __bf16* __restrict__ W1b, const __bf16* __restrict__ W2b,
    const __bf16* __restrict__ WcatA, const __bf16* __restrict__ WcatV,
    const __bf16* __restrict__ Whab, const __bf16* __restrict__ Whvb,
    const __bf16* __restrict__ Wcmb, const float* __restrict__ bcmb,
    float* __restrict__ out) {
  __shared__ float s_imu[2][16 * ST_IV];
  __shared__ float s_vis[2][16 * ST_IV];
  __shared__ float s_Ha [2][16 * ST_H];
  __shared__ float s_Hv [2][16 * ST_H];
  __shared__ float s_ta [2][16 * 32];   // [imu_att(8) | zeros]
  __shared__ float s_tv [2][16 * 32];   // [vis_att(8) | zeros]
  __shared__ float s_tx [2][16 * 32];   // [ha(8) | hv(8) | zeros]

  const int lane = threadIdx.x & 31;
  const int wv   = threadIdx.x >> 5;
  const int ln   = lane & 15;
  const int lh   = (lane >> 4) & 1;
  const int row0 = (blockIdx.x * 2 + wv) * 16;

  float* IMU = s_imu[wv]; float* VIS = s_vis[wv];
  float* HA  = s_Ha[wv];  float* HV  = s_Hv[wv];
  float* TA  = s_ta[wv];  float* TV  = s_tv[wv]; float* TX = s_tx[wv];

  // zero the small tail tiles (their padding columns must be 0)
  for (int i = lane; i < 16 * 32; i += 32) { TA[i] = 0.f; TV[i] = 0.f; TX[i] = 0.f; }

  // ---- enc1: imu = f1 @ W_enc1^T + b_enc1  (K=768, N=64) ----
  {
    v8f acc[4];
#pragma unroll
    for (int nt = 0; nt < 4; ++nt) acc[nt] = v8f_splat(benc1[nt * 16 + ln]);
    const float* arow = f1 + (size_t)(row0 + ln) * 768;
    for (int k = 0; k < 768; k += 32) {
      v16bf a = loadA_gl(arow, k, 768);
#pragma unroll
      for (int nt = 0; nt < 4; ++nt)
        acc[nt] = wmma_bf16(a, loadB(W1b, 768, nt * 16, k), acc[nt]);
    }
#pragma unroll
    for (int nt = 0; nt < 4; ++nt)
#pragma unroll
      for (int r = 0; r < 8; ++r)
        IMU[(r + lh * 8) * ST_IV + nt * 16 + ln] = acc[nt][r];
  }

  // ---- enc2: vis = f2 @ W_enc2^T + b_enc2  (K=600 padded to 608) ----
  {
    v8f acc[4];
#pragma unroll
    for (int nt = 0; nt < 4; ++nt) acc[nt] = v8f_splat(benc2[nt * 16 + ln]);
    const float* arow = f2 + (size_t)(row0 + ln) * 600;
    for (int k = 0; k < 608; k += 32) {
      v16bf a = loadA_gl(arow, k, 600);
#pragma unroll
      for (int nt = 0; nt < 4; ++nt)
        acc[nt] = wmma_bf16(a, loadB(W2b, 608, nt * 16, k), acc[nt]);
    }
#pragma unroll
    for (int nt = 0; nt < 4; ++nt)
#pragma unroll
      for (int r = 0; r < 8; ++r)
        VIS[(r + lh * 8) * ST_IV + nt * 16 + ln] = acc[nt][r];
  }
  __syncthreads();

  // ---- attention gadget: rank-2 bilinear forms + tanh (1 lane / sample) ----
  if (lane < 16) {
    const float* ip = IMU + lane * ST_IV;
    const float* vp = VIS + lane * ST_IV;
    float sii = 0.f, siv = 0.f, svv = 0.f;
    for (int c = 0; c < 64; ++c) {
      float iv = ip[c], vv = vp[c];
      sii += iv * iv; siv += iv * vv; svv += vv * vv;
    }
#pragma unroll
    for (int j = 0; j < 8; ++j) {
      TA[lane * 32 + j] = tanhf((Waa[2 * j] * sii + Waa[2 * j + 1] * siv) * 0.125f);
      TV[lane * 32 + j] = tanhf((Wav[2 * j] * siv + Wav[2 * j + 1] * svv) * 0.125f);
    }
  }
  __syncthreads();

  // ---- H_a = relu(ALPHA * [imu, imu_att] @ [W_a|W_ca]^T)  (K=96, N=32) ----
  {
    v8f a0 = v8f_splat(0.f), a1 = v8f_splat(0.f);
    v16bf afr = loadA_lds(IMU, ST_IV, 0);
    a0 = wmma_bf16(afr, loadB(WcatA, 96, 0, 0), a0);
    a1 = wmma_bf16(afr, loadB(WcatA, 96, 16, 0), a1);
    afr = loadA_lds(IMU, ST_IV, 32);
    a0 = wmma_bf16(afr, loadB(WcatA, 96, 0, 32), a0);
    a1 = wmma_bf16(afr, loadB(WcatA, 96, 16, 32), a1);
    afr = loadA_lds(TA, 32, 0);
    a0 = wmma_bf16(afr, loadB(WcatA, 96, 0, 64), a0);
    a1 = wmma_bf16(afr, loadB(WcatA, 96, 16, 64), a1);
#pragma unroll
    for (int r = 0; r < 8; ++r) {
      HA[(r + lh * 8) * ST_H + ln]      = fmaxf(0.f, ALPHA_F * a0[r]);
      HA[(r + lh * 8) * ST_H + 16 + ln] = fmaxf(0.f, ALPHA_F * a1[r]);
    }
  }
  // ---- H_v = relu((1-A) * [vis, vis_att] @ [W_v|W_cv]^T) ----
  {
    v8f a0 = v8f_splat(0.f), a1 = v8f_splat(0.f);
    v16bf afr = loadA_lds(VIS, ST_IV, 0);
    a0 = wmma_bf16(afr, loadB(WcatV, 96, 0, 0), a0);
    a1 = wmma_bf16(afr, loadB(WcatV, 96, 16, 0), a1);
    afr = loadA_lds(VIS, ST_IV, 32);
    a0 = wmma_bf16(afr, loadB(WcatV, 96, 0, 32), a0);
    a1 = wmma_bf16(afr, loadB(WcatV, 96, 16, 32), a1);
    afr = loadA_lds(TV, 32, 0);
    a0 = wmma_bf16(afr, loadB(WcatV, 96, 0, 64), a0);
    a1 = wmma_bf16(afr, loadB(WcatV, 96, 16, 64), a1);
#pragma unroll
    for (int r = 0; r < 8; ++r) {
      HV[(r + lh * 8) * ST_H + ln]      = fmaxf(0.f, OMA_F * a0[r]);
      HV[(r + lh * 8) * ST_H + 16 + ln] = fmaxf(0.f, OMA_F * a1[r]);
    }
  }
  __syncthreads();

  // ---- ha = H_a @ W_ha^T, hv = H_v @ W_hv^T  (K=32, N=8 padded to 16) ----
  {
    v8f acc = v8f_splat(0.f);
    acc = wmma_bf16(loadA_lds(HA, ST_H, 0), loadB(Whab, 32, 0, 0), acc);
    if (ln < 8) {
#pragma unroll
      for (int r = 0; r < 8; ++r) TX[(r + lh * 8) * 32 + ln] = acc[r];
    }
    v8f accv = v8f_splat(0.f);
    accv = wmma_bf16(loadA_lds(HV, ST_H, 0), loadB(Whvb, 32, 0, 0), accv);
    if (ln < 8) {
#pragma unroll
      for (int r = 0; r < 8; ++r) TX[(r + lh * 8) * 32 + 8 + ln] = accv[r];
    }
  }
  __syncthreads();

  // ---- out = W_combo @ [imu, vis, ha, hv] + b_combo  (K=160, N=32) ----
  {
    v8f o0 = v8f_splat(bcmb[ln]);
    v8f o1 = v8f_splat(bcmb[16 + ln]);
    v16bf afr;
    afr = loadA_lds(IMU, ST_IV, 0);
    o0 = wmma_bf16(afr, loadB(Wcmb, 160, 0, 0), o0);
    o1 = wmma_bf16(afr, loadB(Wcmb, 160, 16, 0), o1);
    afr = loadA_lds(IMU, ST_IV, 32);
    o0 = wmma_bf16(afr, loadB(Wcmb, 160, 0, 32), o0);
    o1 = wmma_bf16(afr, loadB(Wcmb, 160, 16, 32), o1);
    afr = loadA_lds(VIS, ST_IV, 0);
    o0 = wmma_bf16(afr, loadB(Wcmb, 160, 0, 64), o0);
    o1 = wmma_bf16(afr, loadB(Wcmb, 160, 16, 64), o1);
    afr = loadA_lds(VIS, ST_IV, 32);
    o0 = wmma_bf16(afr, loadB(Wcmb, 160, 0, 96), o0);
    o1 = wmma_bf16(afr, loadB(Wcmb, 160, 16, 96), o1);
    afr = loadA_lds(TX, 32, 0);
    o0 = wmma_bf16(afr, loadB(Wcmb, 160, 0, 128), o0);
    o1 = wmma_bf16(afr, loadB(Wcmb, 160, 16, 128), o1);
#pragma unroll
    for (int r = 0; r < 8; ++r) {
      out[(size_t)(row0 + r + lh * 8) * 32 + ln]      = o0[r];
      out[(size_t)(row0 + r + lh * 8) * 32 + 16 + ln] = o1[r];
    }
  }
}

extern "C" void kernel_launch(void* const* d_in, const int* in_sizes, int n_in,
                              void* d_out, int out_size, void* d_ws, size_t ws_size,
                              hipStream_t stream) {
  const float* f1    = (const float*)d_in[0];
  const float* f2    = (const float*)d_in[1];
  const float* Wenc1 = (const float*)d_in[2];
  const float* benc1 = (const float*)d_in[3];
  const float* Wenc2 = (const float*)d_in[4];
  const float* benc2 = (const float*)d_in[5];
  const float* Waa   = (const float*)d_in[6];
  const float* Wav   = (const float*)d_in[7];
  const float* Wa    = (const float*)d_in[8];
  const float* Wv    = (const float*)d_in[9];
  const float* Wca   = (const float*)d_in[10];
  const float* Wcv   = (const float*)d_in[11];
  const float* Wha   = (const float*)d_in[12];
  const float* Whv   = (const float*)d_in[13];
  const float* Wr1   = (const float*)d_in[14];
  const float* br1   = (const float*)d_in[15];
  const float* Wr2   = (const float*)d_in[16];
  const float* br2   = (const float*)d_in[17];

  char* ws = (char*)d_ws;
  float*  Wfold = (float*)(ws + WS_WFOLD);
  __bf16* W1b   = (__bf16*)(ws + WS_W1B);
  __bf16* W2b   = (__bf16*)(ws + WS_W2B);
  __bf16* WcatA = (__bf16*)(ws + WS_WCATA);
  __bf16* WcatV = (__bf16*)(ws + WS_WCATV);
  __bf16* Whab  = (__bf16*)(ws + WS_WHA);
  __bf16* Whvb  = (__bf16*)(ws + WS_WHV);
  __bf16* Wcmb  = (__bf16*)(ws + WS_WCMB);
  float*  bcmb  = (float*)(ws + WS_BCMB);

  const int B = in_sizes[0] / 768;          // 32768
  prep_fold<<<(128 * 160 + 255) / 256, 256, 0, stream>>>(Wr1, Wfold);
  const int packN = 64 * 768 + 64 * 608 + 32 * 96 + 32 * 96 + 16 * 32 + 16 * 32 +
                    32 * 160 + 32;
  prep_pack<<<(packN + 255) / 256, 256, 0, stream>>>(
      Wenc1, Wenc2, Wa, Wv, Wca, Wcv, Wha, Whv, Wr2, br1, br2, Wfold,
      W1b, W2b, WcatA, WcatV, Whab, Whvb, Wcmb, bcmb);
  mica_main<<<B / 32, 64, 0, stream>>>(
      f1, f2, benc1, benc2, Waa, Wav,
      W1b, W2b, WcatA, WcatV, Whab, Whvb, Wcmb, bcmb,
      (float*)d_out);
}
